// GNN_74964359184385
// MI455X (gfx1250) — matrix operand order repaired and checked by reading, not compiled, not run on hardware
//
#include <hip/hip_runtime.h>
#include <hip/hip_bf16.h>
#include <math.h>

#define G_N   20000
#define HIDN  64
#define NH    4
#define E_N   640000
#define B_N   128
#define PD_N  600
#define NEG_SLOPE 0.2f
#define K1    20064   /* G_N + HIDN */
#define N1    1024
#define N2    20000
#define NT1   64      /* N1/16 */
#define NT2   1250    /* N2/16 */
#define KT1   627     /* K1/32 */
#define KT2   32      /* N1/32 */
#define NEDGE (E_N + G_N)

typedef __attribute__((ext_vector_type(16))) __bf16 v16bf;
typedef __attribute__((ext_vector_type(8)))  __bf16 v8bf;
typedef __attribute__((ext_vector_type(8)))  float  v8f;

union ABfrag { v16bf v; struct { v8bf lo, hi; } p; };

#define WMMA_BF16(a, b, c) \
  __builtin_amdgcn_wmma_f32_16x16x32_bf16(false, (a), false, (b), (short)0, (c), false, false)

__device__ __forceinline__ unsigned short f2bf_u16(float f) {   // RNE f32->bf16 bits
  unsigned u = __builtin_bit_cast(unsigned, f);
  return (unsigned short)((u + 0x7FFFu + ((u >> 16) & 1u)) >> 16);
}
__device__ __forceinline__ unsigned enc_f32(float f) {   // order-preserving map
  unsigned u = __builtin_bit_cast(unsigned, f);
  return (u & 0x80000000u) ? ~u : (u | 0x80000000u);
}
__device__ __forceinline__ float dec_f32(unsigned e) {
  unsigned u = (e & 0x80000000u) ? (e ^ 0x80000000u) : ~e;
  return __builtin_bit_cast(float, u);
}
__device__ __forceinline__ float lrelu(float x) { return x > 0.f ? x : NEG_SLOPE * x; }
__device__ __forceinline__ float sigmoidf(float x) { return 1.f / (1.f + __expf(-x)); }

/* ---------------- fills ---------------- */
__global__ void k_fill_f32(float* p, int n, float v) {
  int i = blockIdx.x * blockDim.x + threadIdx.x;
  if (i < n) p[i] = v;
}
__global__ void k_fill_u32(unsigned* p, int n, unsigned v) {
  int i = blockIdx.x * blockDim.x + threadIdx.x;
  if (i < n) p[i] = v;
}

/* ---- pack K x N f32 weights into per-lane WMMA B-fragment bf16 layout ----
   Wp[(kt*Ntiles + nt)*512 + lane*16 + e]  <->  W[k*N + n],
   k = kt*32 + (lane>>4)*8 + (e&7) + (e>>3)*16,  n = nt*16 + (lane&15).       */
__global__ void k_pack_w(const float* __restrict__ W, unsigned short* __restrict__ Wp,
                         int N, int Ntiles, int total) {
  int i = blockIdx.x * blockDim.x + threadIdx.x;
  if (i >= total) return;
  int e    = i & 15;
  int lane = (i >> 4) & 31;
  int tile = i >> 9;
  int nt = tile % Ntiles, kt = tile / Ntiles;
  int k = kt * 32 + (lane >> 4) * 8 + (e & 7) + (e >> 3) * 16;
  int n = nt * 16 + (lane & 15);
  Wp[i] = f2bf_u16(W[(size_t)k * N + n]);
}

/* ---------------- GAT layer 1: per-gene projections ---------------- */
__global__ void k_gat1_pre(const float* __restrict__ x, const int* __restrict__ pos,
                           const float* __restrict__ gemb,
                           const float* __restrict__ W1,     /* 65 x 256 */
                           const float* __restrict__ as1,    /* 4 x 64 */
                           const float* __restrict__ ad1,    /* 4 x 64 */
                           const float* __restrict__ lin1W,  /* 65 x 64 */
                           const float* __restrict__ lin1b,  /* 64 */
                           float* __restrict__ h,            /* G x 256 */
                           float* __restrict__ lin1,         /* G x 64 */
                           float* __restrict__ es, float* __restrict__ ed) /* G x 4 */
{
  int g = blockIdx.x;
  int t = threadIdx.x;                 // 256 threads
  __shared__ float xc[65];
  __shared__ float hrow[256];
  if (t == 0) xc[0] = x[g];
  if (t >= 1 && t < 65) xc[t] = gemb[(size_t)pos[g] * HIDN + (t - 1)];
  __syncthreads();
  float acc = 0.f;
  for (int k = 0; k < 65; ++k) acc += xc[k] * W1[k * 256 + t];
  hrow[t] = acc;
  h[(size_t)g * 256 + t] = acc;
  if (t < HIDN) {
    float a2 = 0.f;
    for (int k = 0; k < 65; ++k) a2 += xc[k] * lin1W[k * HIDN + t];
    lin1[(size_t)g * HIDN + t] = a2 + lin1b[t];
  }
  __syncthreads();
  if (t < 8) {
    int hd = t & 3;
    const float* av = ((t < 4) ? as1 : ad1) + hd * HIDN;
    float s = 0.f;
    for (int c = 0; c < HIDN; ++c) s += hrow[hd * HIDN + c] * av[c];
    ((t < 4) ? es : ed)[g * NH + hd] = s;
  }
}

/* ---------------- edge segment-max (shared by both layers) ---------------- */
__global__ void k_edge_max(const int* __restrict__ ei, const float* __restrict__ es,
                           const float* __restrict__ ed, unsigned* __restrict__ mu)
{
  int i = blockIdx.x * blockDim.x + threadIdx.x;
  if (i >= NEDGE) return;
  int s, d;
  if (i < E_N) { s = ei[i]; d = ei[E_N + i]; } else { s = d = i - E_N; }
  #pragma unroll
  for (int hd = 0; hd < NH; ++hd) {
    float e = lrelu(es[s * NH + hd] + ed[d * NH + hd]);
    atomicMax(&mu[d * NH + hd], enc_f32(e));
  }
}

/* ---------------- layer-1 softmax scatter: wave per edge ---------------- */
__global__ void k_edge_acc1(const int* __restrict__ ei, const float* __restrict__ es,
                            const float* __restrict__ ed, const unsigned* __restrict__ mu,
                            const float* __restrict__ hsrc,
                            float* __restrict__ denom, float* __restrict__ outacc)
{
  int edge = blockIdx.x * 8 + (threadIdx.x >> 5);
  if (edge >= NEDGE) return;
  int lane = threadIdx.x & 31;
  int s, d;
  if (edge < E_N) { s = ei[edge]; d = ei[E_N + edge]; } else { s = d = edge - E_N; }
  int hd = lane >> 3;                                   // 8 lanes per head
  float e  = lrelu(es[s * NH + hd] + ed[d * NH + hd]);
  float ex = __expf(e - dec_f32(mu[d * NH + hd]));
  if ((lane & 7) == 0) atomicAdd(&denom[d * NH + hd], ex);
  int c0 = hd * HIDN + (lane & 7) * 8;
  const float* hr = hsrc + (size_t)s * 256 + c0;
  float* orow = outacc + (size_t)d * 256 + c0;
  #pragma unroll
  for (int j = 0; j < 8; ++j) atomicAdd(&orow[j], ex * hr[j]);
}

/* ---------------- finalize h1 ---------------- */
__global__ void k_fin1(const float* __restrict__ outacc, const float* __restrict__ denom,
                       const float* __restrict__ bias, const float* __restrict__ lin1,
                       float* __restrict__ h1)
{
  int i = blockIdx.x * blockDim.x + threadIdx.x;
  if (i >= G_N * HIDN) return;
  int g = i >> 6, c = i & 63;
  float v = 0.f;
  #pragma unroll
  for (int hd = 0; hd < NH; ++hd)
    v += outacc[(size_t)g * 256 + hd * HIDN + c] / denom[g * NH + hd];
  h1[i] = sigmoidf(0.25f * v + bias[c] + lin1[i]);
}

/* ---------------- GAT layer 2 projections ---------------- */
__global__ void k_prep2(const float* __restrict__ h1, const float* __restrict__ W2c, /* 64x4 */
                        const float* __restrict__ as2, const float* __restrict__ ad2, /* 4 */
                        const float* __restrict__ lin2W, const float* __restrict__ lin2b,
                        float* __restrict__ hh, float* __restrict__ es,
                        float* __restrict__ ed, float* __restrict__ lin2)
{
  int g = blockIdx.x * blockDim.x + threadIdx.x;
  if (g >= G_N) return;
  const float* r = h1 + (size_t)g * HIDN;
  float acc[NH] = {0.f, 0.f, 0.f, 0.f};
  float l = 0.f;
  for (int c = 0; c < HIDN; ++c) {
    float xv = r[c];
    #pragma unroll
    for (int hd = 0; hd < NH; ++hd) acc[hd] += xv * W2c[c * NH + hd];
    l += xv * lin2W[c];
  }
  #pragma unroll
  for (int hd = 0; hd < NH; ++hd) {
    hh[g * NH + hd] = acc[hd];
    es[g * NH + hd] = acc[hd] * as2[hd];
    ed[g * NH + hd] = acc[hd] * ad2[hd];
  }
  lin2[g] = l + lin2b[0];
}

/* ---------------- layer-2 softmax scatter: thread per edge ---------------- */
__global__ void k_edge_acc2(const int* __restrict__ ei, const float* __restrict__ es,
                            const float* __restrict__ ed, const unsigned* __restrict__ mu,
                            const float* __restrict__ hh,
                            float* __restrict__ denom, float* __restrict__ outacc)
{
  int i = blockIdx.x * blockDim.x + threadIdx.x;
  if (i >= NEDGE) return;
  int s, d;
  if (i < E_N) { s = ei[i]; d = ei[E_N + i]; } else { s = d = i - E_N; }
  #pragma unroll
  for (int hd = 0; hd < NH; ++hd) {
    float e  = lrelu(es[s * NH + hd] + ed[d * NH + hd]);
    float ex = __expf(e - dec_f32(mu[d * NH + hd]));
    atomicAdd(&denom[d * NH + hd], ex);
    atomicAdd(&outacc[d * NH + hd], ex * hh[s * NH + hd]);
  }
}

__global__ void k_fin2(const float* __restrict__ out2, const float* __restrict__ den2,
                       const float* __restrict__ b2, const float* __restrict__ lin2,
                       float* __restrict__ feat)
{
  int g = blockIdx.x * blockDim.x + threadIdx.x;
  if (g >= G_N) return;
  float v = 0.f;
  #pragma unroll
  for (int hd = 0; hd < NH; ++hd) v += out2[g * NH + hd] / den2[g * NH + hd];
  feat[g] = sigmoidf(0.25f * v + b2[0] + lin2[g]);
}

/* ---------------- pert MLP: block per batch row ---------------- */
__global__ void k_pert(const float* __restrict__ pert, const float* __restrict__ W1,
                       const float* __restrict__ b1, const float* __restrict__ W2,
                       const float* __restrict__ b2, float* __restrict__ p)
{
  int b = blockIdx.x, t = threadIdx.x;   // 128 threads
  __shared__ float s1[128];
  const float* pr = pert + (size_t)b * PD_N;
  float acc = b1[t];
  for (int k = 0; k < PD_N; ++k) acc += pr[k] * W1[k * 128 + t];
  s1[t] = sigmoidf(acc);
  __syncthreads();
  if (t < HIDN) {
    float a2 = b2[t];
    for (int k = 0; k < 128; ++k) a2 += s1[k] * W2[k * HIDN + t];
    p[b * HIDN + t] = a2;
  }
}

/* ---------------- build inp = [feat+ctrl | p] directly in bf16 ---------------- */
__global__ void k_build(const float* __restrict__ feat, const float* __restrict__ ctrl,
                        const float* __restrict__ p, unsigned short* __restrict__ A16)
{
  int i = blockIdx.x * blockDim.x + threadIdx.x;
  if (i >= B_N * K1) return;
  int b = i / K1, j = i % K1;
  float v = (j < G_N) ? (feat[j] + ctrl[(size_t)b * G_N + j]) : p[b * HIDN + (j - G_N)];
  A16[i] = f2bf_u16(v);
}

/* ---- WMMA GEMM 1: S16 = bf16(sigmoid(A16 @ W1p + b1)), 128x1024.
   2 M-tiles x 1 N-tile per wave; double-buffered fragment loads. ---- */
__global__ void k_gemm1(const unsigned short* __restrict__ A16, /* 128 x K1 bf16 */
                        const unsigned short* __restrict__ Wp,  /* packed fragments */
                        const float* __restrict__ bias,
                        unsigned short* __restrict__ S16)
{
  int wv   = (blockIdx.x << 3) + (threadIdx.x >> 5);   // 256 waves total
  int lane = threadIdx.x & 31;
  int mtp = wv >> 6, nt = wv & 63;
  int m0 = mtp << 5, n0 = nt << 4;
  int m = lane & 15, hi = lane >> 4, n = lane & 15;
  v8f acc0 = {}, acc1 = {};
  const v8bf*  a0 = (const v8bf*)(A16 + (size_t)(m0 + m) * K1) + hi;
  const v8bf*  a1 = (const v8bf*)(A16 + (size_t)(m0 + 16 + m) * K1) + hi;
  const v16bf* bp = (const v16bf*)Wp + (size_t)nt * 32 + lane;

  ABfrag f0c, f1c; v16bf bc;
  f0c.p.lo = a0[0]; f0c.p.hi = a0[2];
  f1c.p.lo = a1[0]; f1c.p.hi = a1[2];
  bc = *bp;
  #pragma unroll 2
  for (int kt = 0; kt < KT1 - 1; ++kt) {
    ABfrag f0n, f1n; v16bf bn;
    a0 += 4; a1 += 4; bp += (size_t)NT1 * 32;
    f0n.p.lo = a0[0]; f0n.p.hi = a0[2];
    f1n.p.lo = a1[0]; f1n.p.hi = a1[2];
    bn = *bp;
    acc0 = WMMA_BF16(f0c.v, bc, acc0);
    acc1 = WMMA_BF16(f1c.v, bc, acc1);
    f0c = f0n; f1c = f1n; bc = bn;
  }
  acc0 = WMMA_BF16(f0c.v, bc, acc0);
  acc1 = WMMA_BF16(f1c.v, bc, acc1);

  #pragma unroll
  for (int r = 0; r < 8; ++r) {
    int mr = (lane < 16) ? r : r + 8;
    int nn = n0 + n;
    float bv = bias[nn];
    S16[(size_t)(m0 + mr)      * N1 + nn] = f2bf_u16(sigmoidf(acc0[r] + bv));
    S16[(size_t)(m0 + 16 + mr) * N1 + nn] = f2bf_u16(sigmoidf(acc1[r] + bv));
  }
}

/* ---- WMMA GEMM 2: out = S16 @ W2p + b2, 128x20000 f32.
   4 M-tiles x 1 N-tile per wave (one B-fragment feeds 4 WMMAs);
   double-buffered. 2500 waves = 625 blocks x 4 waves. ---- */
__global__ void k_gemm2(const unsigned short* __restrict__ A16, /* 128 x 1024 bf16 */
                        const unsigned short* __restrict__ Wp,
                        const float* __restrict__ bias,
                        float* __restrict__ out)
{
  int wv   = (blockIdx.x << 2) + (threadIdx.x >> 5);   // 2500 waves total
  int lane = threadIdx.x & 31;
  int mtp = wv & 1, nt = wv >> 1;                      // nt: 0..1249
  int m0 = mtp << 6, n0 = nt << 4;
  int m = lane & 15, hi = lane >> 4, n = lane & 15;
  v8f acc[4] = {{}, {}, {}, {}};
  const v8bf* ap[4];
  #pragma unroll
  for (int t = 0; t < 4; ++t)
    ap[t] = (const v8bf*)(A16 + (size_t)(m0 + 16 * t + m) * N1) + hi;
  const v16bf* bp = (const v16bf*)Wp + (size_t)nt * 32 + lane;

  ABfrag fc[4]; v16bf bc;
  #pragma unroll
  for (int t = 0; t < 4; ++t) { fc[t].p.lo = ap[t][0]; fc[t].p.hi = ap[t][2]; }
  bc = *bp;
  #pragma unroll 2
  for (int kt = 0; kt < KT2 - 1; ++kt) {
    ABfrag fn[4]; v16bf bn;
    #pragma unroll
    for (int t = 0; t < 4; ++t) {
      ap[t] += 4;
      fn[t].p.lo = ap[t][0]; fn[t].p.hi = ap[t][2];
    }
    bp += (size_t)NT2 * 32;
    bn = *bp;
    #pragma unroll
    for (int t = 0; t < 4; ++t) acc[t] = WMMA_BF16(fc[t].v, bc, acc[t]);
    #pragma unroll
    for (int t = 0; t < 4; ++t) fc[t] = fn[t];
    bc = bn;
  }
  #pragma unroll
  for (int t = 0; t < 4; ++t) acc[t] = WMMA_BF16(fc[t].v, bc, acc[t]);

  #pragma unroll
  for (int t = 0; t < 4; ++t) {
    #pragma unroll
    for (int r = 0; r < 8; ++r) {
      int mr = (lane < 16) ? r : r + 8;
      int nn = n0 + n;
      out[(size_t)(m0 + 16 * t + mr) * N2 + nn] = acc[t][r] + bias[nn];
    }
  }
}

/* ---------------- host ---------------- */
static inline int cdiv(int a, int b) { return (a + b - 1) / b; }

extern "C" void kernel_launch(void* const* d_in, const int* in_sizes, int n_in,
                              void* d_out, int out_size, void* d_ws, size_t ws_size,
                              hipStream_t stream) {
  (void)in_sizes; (void)n_in; (void)out_size; (void)ws_size;
  const float* x        = (const float*)d_in[0];
  const int*   ei       = (const int*)  d_in[1];
  const int*   pos      = (const int*)  d_in[2];
  const float* ctrl     = (const float*)d_in[3];
  const float* pert     = (const float*)d_in[4];
  const float* gemb     = (const float*)d_in[5];
  const float* conv1_W  = (const float*)d_in[6];
  const float* conv1_as = (const float*)d_in[7];
  const float* conv1_ad = (const float*)d_in[8];
  const float* conv1_b  = (const float*)d_in[9];
  const float* lin1_W   = (const float*)d_in[10];
  const float* lin1_b   = (const float*)d_in[11];
  const float* conv2_W  = (const float*)d_in[12];
  const float* conv2_as = (const float*)d_in[13];
  const float* conv2_ad = (const float*)d_in[14];
  const float* conv2_b  = (const float*)d_in[15];
  const float* lin2_W   = (const float*)d_in[16];
  const float* lin2_b   = (const float*)d_in[17];
  const float* pert_W1  = (const float*)d_in[18];
  const float* pert_b1  = (const float*)d_in[19];
  const float* pert_W2  = (const float*)d_in[20];
  const float* pert_b2  = (const float*)d_in[21];
  const float* pred_W1  = (const float*)d_in[22];
  const float* pred_b1  = (const float*)d_in[23];
  const float* pred_W2  = (const float*)d_in[24];
  const float* pred_b2  = (const float*)d_in[25];
  float* out = (float*)d_out;

  /* workspace layout (f32 units; all sub-arrays 16B-aligned) */
  float* ws = (float*)d_ws;
  size_t o = 0;
  float*    h      = ws + o; o += (size_t)G_N * 256;
  float*    lin1   = ws + o; o += (size_t)G_N * HIDN;
  float*    es1    = ws + o; o += (size_t)G_N * NH;
  float*    ed1    = ws + o; o += (size_t)G_N * NH;
  unsigned* mu1    = (unsigned*)(ws + o); o += (size_t)G_N * NH;
  float*    den1   = ws + o; o += (size_t)G_N * NH;
  float*    oacc1  = ws + o; o += (size_t)G_N * 256;
  float*    h1     = ws + o; o += (size_t)G_N * HIDN;
  float*    hh2    = ws + o; o += (size_t)G_N * NH;
  float*    es2    = ws + o; o += (size_t)G_N * NH;
  float*    ed2    = ws + o; o += (size_t)G_N * NH;
  unsigned* mu2    = (unsigned*)(ws + o); o += (size_t)G_N * NH;
  float*    den2   = ws + o; o += (size_t)G_N * NH;
  float*    oacc2  = ws + o; o += (size_t)G_N * NH;
  float*    lin2   = ws + o; o += (size_t)G_N;
  float*    feat   = ws + o; o += (size_t)G_N;
  float*    pvec   = ws + o; o += (size_t)B_N * HIDN;
  const int W1P_TOT = KT1 * NT1 * 512;                  /* 20,545,536 bf16 */
  const int W2P_TOT = KT2 * NT2 * 512;                  /* 20,480,000 bf16 */
  unsigned short* W1p = (unsigned short*)(ws + o); o += (size_t)W1P_TOT / 2;
  unsigned short* W2p = (unsigned short*)(ws + o); o += (size_t)W2P_TOT / 2;
  unsigned short* A16 = (unsigned short*)(ws + o); o += (size_t)B_N * K1 / 2;
  unsigned short* S16 = (unsigned short*)(ws + o); o += (size_t)B_N * N1 / 2;

  const unsigned NEG_INF_ENC = 0x007FFFFFu;   /* enc(-inf) */

  /* weight packing (independent; launched first) */
  k_pack_w<<<cdiv(W1P_TOT, 256), 256, 0, stream>>>(pred_W1, W1p, N1, NT1, W1P_TOT);
  k_pack_w<<<cdiv(W2P_TOT, 256), 256, 0, stream>>>(pred_W2, W2p, N2, NT2, W2P_TOT);

  /* GAT layer 1 */
  k_gat1_pre<<<G_N, 256, 0, stream>>>(x, pos, gemb, conv1_W, conv1_as, conv1_ad,
                                      lin1_W, lin1_b, h, lin1, es1, ed1);
  k_fill_u32<<<cdiv(G_N * NH, 256), 256, 0, stream>>>(mu1, G_N * NH, NEG_INF_ENC);
  k_fill_f32<<<cdiv(G_N * NH, 256), 256, 0, stream>>>(den1, G_N * NH, 0.f);
  k_fill_f32<<<cdiv(G_N * 256, 256), 256, 0, stream>>>(oacc1, G_N * 256, 0.f);
  k_edge_max<<<cdiv(NEDGE, 256), 256, 0, stream>>>(ei, es1, ed1, mu1);
  k_edge_acc1<<<cdiv(NEDGE, 8), 256, 0, stream>>>(ei, es1, ed1, mu1, h, den1, oacc1);
  k_fin1<<<cdiv(G_N * HIDN, 256), 256, 0, stream>>>(oacc1, den1, conv1_b, lin1, h1);

  /* GAT layer 2 */
  k_prep2<<<cdiv(G_N, 256), 256, 0, stream>>>(h1, conv2_W, conv2_as, conv2_ad,
                                              lin2_W, lin2_b, hh2, es2, ed2, lin2);
  k_fill_u32<<<cdiv(G_N * NH, 256), 256, 0, stream>>>(mu2, G_N * NH, NEG_INF_ENC);
  k_fill_f32<<<cdiv(G_N * NH, 256), 256, 0, stream>>>(den2, G_N * NH, 0.f);
  k_fill_f32<<<cdiv(G_N * NH, 256), 256, 0, stream>>>(oacc2, G_N * NH, 0.f);
  k_edge_max<<<cdiv(NEDGE, 256), 256, 0, stream>>>(ei, es2, ed2, mu2);
  k_edge_acc2<<<cdiv(NEDGE, 256), 256, 0, stream>>>(ei, es2, ed2, mu2, hh2, den2, oacc2);
  k_fin2<<<cdiv(G_N, 256), 256, 0, stream>>>(oacc2, den2, conv2_b, lin2, feat);

  /* pert MLP + bf16 input assembly */
  k_pert<<<B_N, 128, 0, stream>>>(pert, pert_W1, pert_b1, pert_W2, pert_b2, pvec);
  k_build<<<cdiv(B_N * K1, 256), 256, 0, stream>>>(feat, ctrl, pvec, A16);

  /* prediction MLP via bf16 WMMA (f32 accumulate), double-buffered hot loops */
  k_gemm1<<<32, 256, 0, stream>>>(A16, W1p, pred_b1, S16);     /* 256 waves  */
  k_gemm2<<<625, 128, 0, stream>>>(S16, W2p, pred_b2, out);    /* 2500 waves */
}
